// STGNNBlock_18880676233980
// MI455X (gfx1250) — matrix operand order repaired and checked by reading, not compiled, not run on hardware
//
#include <hip/hip_runtime.h>
#include <hip/hip_bf16.h>

typedef __attribute__((ext_vector_type(16))) _Float16 v16h;
typedef __attribute__((ext_vector_type(8)))  _Float16 h8v;
typedef __attribute__((ext_vector_type(8)))  float    v8f;

#define B_   64
#define N_   3
#define T_   4096
#define D_   64
#define H_   128
#define MALL (B_ * N_ * T_)   // 786432 rows

// ---------------- WMMA fragment loaders (layouts per CDNA5 ISA 7.12.2) ----------
// All loaders issue 16-byte LDS accesses (ds_load_b128), no scalar u16 traffic.

// A: 16x32 f16 tile, row-major a[row*lda + k]. Lane L holds row L%16;
// halves 0..7 -> K kbase..kbase+7, halves 8..15 -> K kbase+16..kbase+23,
// kbase = 8*(L>=16).
__device__ __forceinline__ v16h load_a_frag(const _Float16* a, int lda, int lane) {
  const int row   = lane & 15;
  const int kbase = (lane >> 4) << 3;     // 0 or 8
  const _Float16* p = a + row * lda + kbase;
  h8v x0 = *(const h8v*)(p);
  h8v x1 = *(const h8v*)(p + 16);
  return __builtin_shufflevector(x0, x1, 0, 1, 2, 3, 4, 5, 6, 7,
                                 8, 9, 10, 11, 12, 13, 14, 15);
}

// B: 32x16 f16 tile from N-major storage b[n*ldn + k] (torch (out,in) layout).
// Lane L holds column L%16; halves 0..7 -> K kbase..kbase+7, halves 8..15 -> K kbase+16..kbase+23,
// kbase = 8*(L>=16).
__device__ __forceinline__ v16h load_b_nmajor(const _Float16* b, int ldn, int lane) {
  const _Float16* p = b + (lane & 15) * ldn + ((lane >> 4) << 3);
  h8v x0 = *(const h8v*)(p);
  h8v x1 = *(const h8v*)(p + 16);
  return __builtin_shufflevector(x0, x1, 0, 1, 2, 3, 4, 5, 6, 7,
                                 8, 9, 10, 11, 12, 13, 14, 15);
}

// ---------------- K0: weight prep (fp32 -> f16, W transposed, score vectors) ----

__global__ void k0_prep(const float* __restrict__ gat_W, const float* __restrict__ gat_b,
                        const float* __restrict__ a_src, const float* __restrict__ a_dst,
                        const float* __restrict__ W_ih,  const float* __restrict__ W_hh,
                        _Float16* __restrict__ wgatT16,  // (H, D) N-major
                        _Float16* __restrict__ wih16,    // (3H, H) as-is (N-major)
                        _Float16* __restrict__ whh16,
                        float* __restrict__ wsrc, float* __restrict__ wdst,
                        float* __restrict__ cc) {
  const int tid = threadIdx.x;
  for (int i = tid; i < D_ * H_; i += 256) {
    int d = i / H_, h = i % H_;
    wgatT16[h * D_ + d] = (_Float16)gat_W[i];
  }
  for (int i = tid; i < 3 * H_ * H_; i += 256) {
    wih16[i] = (_Float16)W_ih[i];
    whh16[i] = (_Float16)W_hh[i];
  }
  if (tid < D_) {
    float a = 0.f, b = 0.f;
    for (int h = 0; h < H_; ++h) {
      float w = gat_W[tid * H_ + h];
      a += w * a_src[h];
      b += w * a_dst[h];
    }
    wsrc[tid] = a; wdst[tid] = b;
  }
  if (tid == D_) {
    float a = 0.f, b = 0.f;
    for (int h = 0; h < H_; ++h) { a += gat_b[h] * a_src[h]; b += gat_b[h] * a_dst[h]; }
    cc[0] = a; cc[1] = b;
  }
}

// ---------------- K1: h = x @ W + b  (WMMA, 128-row x 128-col tiles, f16 out) --

__global__ void __launch_bounds__(256) k1_gat_h(const float* __restrict__ x,
                                                const _Float16* __restrict__ wT16,
                                                const float* __restrict__ gat_b,
                                                _Float16* __restrict__ h16) {
  __shared__ __align__(32) _Float16 xs[128 * D_];   // 16 KB
  __shared__ __align__(32) _Float16 ws[H_ * D_];    // 16 KB, N-major (H,D)
  const int tid = threadIdx.x;
  const long long m0 = (long long)blockIdx.x * 128;

  for (int i = tid; i < 128 * D_; i += 256)
    xs[i] = (_Float16)x[m0 * D_ + i];
  for (int i = tid; i < (H_ * D_) / 8; i += 256)
    ((h8v*)ws)[i] = ((const h8v*)wT16)[i];
  __syncthreads();

  const int wave = tid >> 5, lane = tid & 31;
  const int mt = wave;                       // 8 waves -> 8 M-subtiles of 16 rows
  for (int nt = 0; nt < 8; ++nt) {
    v8f acc = {};
#pragma unroll
    for (int kk = 0; kk < 2; ++kk) {         // K = 64 = 2 x 32
      v16h a = load_a_frag(xs + (mt * 16) * D_ + kk * 32, D_, lane);
      v16h b = load_b_nmajor(ws + (nt * 16) * D_ + kk * 32, D_, lane);
      acc = __builtin_amdgcn_wmma_f32_16x16x32_f16(false, a, false, b, (short)0, acc,
                                                   false, false);
    }
    const int col   = nt * 16 + (lane & 15);
    const float bias = gat_b[col];
    const int rbase = (lane >> 4) * 8;
#pragma unroll
    for (int i = 0; i < 8; ++i) {
      long long row = m0 + mt * 16 + rbase + i;
      h16[row * H_ + col] = (_Float16)(acc[i] + bias);
    }
  }
}

// ---------------- K1b: s_src/s_dst = x . (W a) + b.a  (one wave per row) -------

__global__ void __launch_bounds__(256) k1b_scores(const float* __restrict__ x,
                                                  const float* __restrict__ wsrc,
                                                  const float* __restrict__ wdst,
                                                  const float* __restrict__ cc,
                                                  float* __restrict__ ssrc,
                                                  float* __restrict__ sdst) {
  const int lane = threadIdx.x & 31;
  const long long m = (long long)blockIdx.x * 8 + (threadIdx.x >> 5);
  if (m >= (long long)MALL) return;
  const float* xr = x + m * D_;
  float v0 = xr[lane], v1 = xr[lane + 32];
  float a = v0 * wsrc[lane] + v1 * wsrc[lane + 32];
  float b = v0 * wdst[lane] + v1 * wdst[lane + 32];
#pragma unroll
  for (int off = 16; off > 0; off >>= 1) {
    a += __shfl_down(a, off, 32);
    b += __shfl_down(b, off, 32);
  }
  if (lane == 0) { ssrc[m] = a + cc[0]; sdst[m] = b + cc[1]; }
}

// ---------------- K2: softmax attention (N=3) + ELU + spatial LN ---------------
// One block handles 8 consecutive timesteps of one batch element.

#define K2_TCHUNK 8

__global__ void __launch_bounds__(128) k2_attn(const _Float16* __restrict__ h16,
                                               const float* __restrict__ ssrc,
                                               const float* __restrict__ sdst,
                                               const float* __restrict__ bias,
                                               const float* __restrict__ sng,
                                               const float* __restrict__ snb,
                                               _Float16* __restrict__ hsp16) {
  const int c = threadIdx.x;                 // channel 0..127
  const long long chunk = blockIdx.x;        // (b*T + t0) / K2_TCHUNK
  const long long b  = chunk / (T_ / K2_TCHUNK);
  const long long t0 = (chunk % (T_ / K2_TCHUNK)) * K2_TCHUNK;
  const float gw = sng[c], gb = snb[c];

  __shared__ float red[N_][2][4];

  for (int tt = 0; tt < K2_TCHUNK; ++tt) {
    const long long t = t0 + tt;
    float ss[N_], sd[N_], hv[N_];
#pragma unroll
    for (int i = 0; i < N_; ++i) {
      long long m = (b * N_ + i) * T_ + t;
      ss[i] = ssrc[m]; sd[i] = sdst[m];
      hv[i] = (float)h16[m * H_ + c];
    }

    float outv[N_];
#pragma unroll
    for (int i = 0; i < N_; ++i) {
      float sc[N_], mx = -1e30f;
#pragma unroll
      for (int j = 0; j < N_; ++j) {
        float s = ss[i] + sd[j];
        s = (s > 0.f) ? s : 0.2f * s;        // leaky_relu(0.2)
        s += bias[i * N_ + j];
        sc[j] = s; mx = fmaxf(mx, s);
      }
      float den = 0.f, acc = 0.f;
#pragma unroll
      for (int j = 0; j < N_; ++j) { sc[j] = __expf(sc[j] - mx); den += sc[j]; }
#pragma unroll
      for (int j = 0; j < N_; ++j) acc += sc[j] * hv[j];
      acc /= den;
      outv[i] = (acc > 0.f) ? acc : (__expf(acc) - 1.f);   // ELU
    }

    // spatial LayerNorm over channels (3 rows at once); 4 wave32s per block
#pragma unroll
    for (int i = 0; i < N_; ++i) {
      float s1 = outv[i], s2 = outv[i] * outv[i];
#pragma unroll
      for (int off = 16; off > 0; off >>= 1) {
        s1 += __shfl_down(s1, off, 32);
        s2 += __shfl_down(s2, off, 32);
      }
      if ((threadIdx.x & 31) == 0) {
        red[i][0][threadIdx.x >> 5] = s1;
        red[i][1][threadIdx.x >> 5] = s2;
      }
    }
    __syncthreads();
#pragma unroll
    for (int i = 0; i < N_; ++i) {
      float s1 = red[i][0][0] + red[i][0][1] + red[i][0][2] + red[i][0][3];
      float s2 = red[i][1][0] + red[i][1][1] + red[i][1][2] + red[i][1][3];
      float mean = s1 * (1.f / H_);
      float var  = s2 * (1.f / H_) - mean * mean;
      float rstd = rsqrtf(var + 1e-5f);
      hsp16[((b * N_ + i) * T_ + t) * H_ + c] =
          (_Float16)((outv[i] - mean) * rstd * gw + gb);
    }
    __syncthreads();                         // red reuse next iteration
  }
}

// ---------------- K3: gx = h_sp @ W_ih^T + b_ih  (WMMA, N chunked, f16 out) ----

__global__ void __launch_bounds__(256) k3_gx(const _Float16* __restrict__ hsp16,
                                             const _Float16* __restrict__ wih16,
                                             const float* __restrict__ b_ih,
                                             _Float16* __restrict__ gx16) {
  __shared__ __align__(32) _Float16 hs[128 * H_];   // 32 KB
  __shared__ __align__(32) _Float16 wc[128 * H_];   // 32 KB (one 128-gate chunk)
  const int tid = threadIdx.x;
  const long long m0 = (long long)blockIdx.x * 128;

  for (int i = tid; i < (128 * H_) / 8; i += 256)
    ((h8v*)hs)[i] = ((const h8v*)(hsp16 + m0 * H_))[i];
  const int wave = tid >> 5, lane = tid & 31;
  const int mt = wave;

  for (int g = 0; g < 3; ++g) {
    __syncthreads();                                 // prior compute / hs load done
    for (int i = tid; i < (128 * H_) / 8; i += 256)
      ((h8v*)wc)[i] = ((const h8v*)(wih16 + g * 128 * H_))[i];
    __syncthreads();
    for (int nt = 0; nt < 8; ++nt) {
      v8f acc = {};
#pragma unroll
      for (int kk = 0; kk < 4; ++kk) {               // K = 128 = 4 x 32
        v16h a = load_a_frag(hs + (mt * 16) * H_ + kk * 32, H_, lane);
        v16h b = load_b_nmajor(wc + (nt * 16) * H_ + kk * 32, H_, lane);
        acc = __builtin_amdgcn_wmma_f32_16x16x32_f16(false, a, false, b, (short)0, acc,
                                                     false, false);
      }
      const int gcol = g * 128 + nt * 16 + (lane & 15);
      const float bias = b_ih[gcol];
      const int rbase = (lane >> 4) * 8;
#pragma unroll
      for (int i = 0; i < 8; ++i) {
        long long row = m0 + mt * 16 + rbase + i;
        gx16[row * 384 + gcol] = (_Float16)(acc[i] + bias);
      }
    }
  }
}

// ---------------- K4: GRU scan over T + fused temporal LN ----------------------
// 12 workgroups x 16 rows; W_hh, h-state, gate buffer all LDS-resident (~140 KB).

__global__ void __launch_bounds__(256, 1) k4_scan(const _Float16* __restrict__ gx16,
                                                  const _Float16* __restrict__ whh16,
                                                  const float* __restrict__ b_hh,
                                                  const float* __restrict__ tng,
                                                  const float* __restrict__ tnb,
                                                  float* __restrict__ out) {
  __shared__ __align__(32) _Float16 whh[384 * H_];   // 96 KB
  __shared__ __align__(32) _Float16 hf16[16 * H_];   // 4 KB  h state (matmul operand)
  __shared__ float    hf32[16 * H_];                 // 8 KB  h state (f32, exact)
  __shared__ float    gh[16 * 384];                  // 24 KB per-step matmul result
  __shared__ float    bh[384];
  __shared__ float    gn[H_], bn[H_];
  __shared__ float    red1[16][17], red2[16][17];
  __shared__ float    mv[16][2];

  const int tid = threadIdx.x, wave = tid >> 5, lane = tid & 31;
  const long long m0 = (long long)blockIdx.x * 16;

  for (int i = tid; i < (384 * H_) / 8; i += 256)
    ((h8v*)whh)[i] = ((const h8v*)whh16)[i];
  for (int i = tid; i < 16 * H_; i += 256) { hf16[i] = (_Float16)0.f; hf32[i] = 0.f; }
  for (int i = tid; i < 384; i += 256) bh[i] = b_hh[i];
  for (int i = tid; i < H_;  i += 256) { gn[i] = tng[i]; bn[i] = tnb[i]; }
  __syncthreads();

  const int r = tid >> 4;                   // row 0..15
  const int j = tid & 15;                   // 16 threads per row

  for (int t = 0; t < T_; ++t) {
    // -- matmul phase: gh = h @ W_hh^T; 24 N-tiles, 3 per wave --
#pragma unroll
    for (int q = 0; q < 3; ++q) {
      const int nt = wave * 3 + q;
      v8f acc = {};
#pragma unroll
      for (int kk = 0; kk < 4; ++kk) {
        v16h a = load_a_frag(hf16 + kk * 32, H_, lane);
        v16h b = load_b_nmajor(whh + (nt * 16) * H_ + kk * 32, H_, lane);
        acc = __builtin_amdgcn_wmma_f32_16x16x32_f16(false, a, false, b, (short)0, acc,
                                                     false, false);
      }
      const int col = nt * 16 + (lane & 15);
      const int rbase = (lane >> 4) * 8;
#pragma unroll
      for (int i = 0; i < 8; ++i) gh[(rbase + i) * 384 + col] = acc[i];
    }
    __syncthreads();                                   // gh ready; hf16 reads done

    // -- gate phase: thread handles row r, channels c = j + 16k --
    float hn[8], s1 = 0.f, s2 = 0.f;
    const _Float16* gxr = gx16 + ((m0 + r) * T_ + t) * 384;
#pragma unroll
    for (int k = 0; k < 8; ++k) {
      const int c = j + 16 * k;
      const float xr_ = (float)gxr[c];
      const float xz_ = (float)gxr[c + 128];
      const float xn_ = (float)gxr[c + 256];
      const float hr_ = gh[r * 384 + c]       + bh[c];
      const float hz_ = gh[r * 384 + c + 128] + bh[c + 128];
      const float hn_ = gh[r * 384 + c + 256] + bh[c + 256];
      const float rg = 1.f / (1.f + __expf(-(xr_ + hr_)));
      const float zg = 1.f / (1.f + __expf(-(xz_ + hz_)));
      const float ng = tanhf(xn_ + rg * hn_);
      const float hp = hf32[r * H_ + c];
      const float hv = (1.f - zg) * ng + zg * hp;
      hn[k] = hv; s1 += hv; s2 += hv * hv;
      hf32[r * H_ + c] = hv;
      hf16[r * H_ + c] = (_Float16)hv;
    }
    red1[r][j] = s1; red2[r][j] = s2;
    if (t + 1 < T_) __builtin_prefetch(gxr + 384, 0, 0);   // global_prefetch_b8
    __syncthreads();                                   // partials + new h ready

    if (j == 0) {
      float a = 0.f, bq = 0.f;
#pragma unroll
      for (int q = 0; q < 16; ++q) { a += red1[r][q]; bq += red2[r][q]; }
      const float mean = a * (1.f / H_);
      const float var  = bq * (1.f / H_) - mean * mean;
      mv[r][0] = mean; mv[r][1] = rsqrtf(var + 1e-5f);
    }
    __syncthreads();

    const float mean = mv[r][0], rstd = mv[r][1];
    float* orow = out + ((m0 + r) * T_ + t) * H_;
#pragma unroll
    for (int k = 0; k < 8; ++k) {
      const int c = j + 16 * k;
      orow[c] = (hn[k] - mean) * rstd * gn[c] + bn[c];
    }
  }
}

// ---------------- launch ------------------------------------------------------

extern "C" void kernel_launch(void* const* d_in, const int* in_sizes, int n_in,
                              void* d_out, int out_size, void* d_ws, size_t ws_size,
                              hipStream_t stream) {
  const float* x       = (const float*)d_in[0];
  const float* gat_W   = (const float*)d_in[1];
  const float* gat_b   = (const float*)d_in[2];
  const float* a_src   = (const float*)d_in[3];
  const float* a_dst   = (const float*)d_in[4];
  const float* attn_b  = (const float*)d_in[5];
  const float* sn_g    = (const float*)d_in[6];
  const float* sn_b    = (const float*)d_in[7];
  const float* W_ih    = (const float*)d_in[8];
  const float* W_hh    = (const float*)d_in[9];
  const float* b_ih    = (const float*)d_in[10];
  const float* b_hh    = (const float*)d_in[11];
  const float* tn_g    = (const float*)d_in[12];
  const float* tn_b    = (const float*)d_in[13];
  float* out = (float*)d_out;

  char* ws = (char*)d_ws;
  size_t off = 0;
  auto alloc = [&](size_t bytes) { char* p = ws + off; off = (off + bytes + 255) & ~(size_t)255; return p; };

  _Float16*  h16    = (_Float16*)alloc((size_t)MALL * H_ * sizeof(_Float16));  // 0.20 GB
  _Float16*  hsp16  = (_Float16*)alloc((size_t)MALL * H_ * sizeof(_Float16));  // 0.20 GB
  _Float16*  gx16   = (_Float16*)alloc((size_t)MALL * 384 * sizeof(_Float16)); // 0.60 GB
  float*     ssrc   = (float*)alloc((size_t)MALL * sizeof(float));
  float*     sdst   = (float*)alloc((size_t)MALL * sizeof(float));
  _Float16*  wgatT16 = (_Float16*)alloc((size_t)D_ * H_ * sizeof(_Float16));
  _Float16*  wih16   = (_Float16*)alloc((size_t)3 * H_ * H_ * sizeof(_Float16));
  _Float16*  whh16   = (_Float16*)alloc((size_t)3 * H_ * H_ * sizeof(_Float16));
  float*     wsrc    = (float*)alloc(D_ * sizeof(float));
  float*     wdst    = (float*)alloc(D_ * sizeof(float));
  float*     cc      = (float*)alloc(2 * sizeof(float));

  k0_prep<<<1, 256, 0, stream>>>(gat_W, gat_b, a_src, a_dst, W_ih, W_hh,
                                 wgatT16, wih16, whh16, wsrc, wdst, cc);
  k1_gat_h<<<MALL / 128, 256, 0, stream>>>(x, wgatT16, gat_b, h16);
  k1b_scores<<<MALL / 8, 256, 0, stream>>>(x, wsrc, wdst, cc, ssrc, sdst);
  k2_attn<<<(B_ * T_) / K2_TCHUNK, 128, 0, stream>>>(h16, ssrc, sdst, attn_b,
                                                     sn_g, sn_b, hsp16);
  k3_gx<<<MALL / 128, 256, 0, stream>>>(hsp16, wih16, b_ih, gx16);
  k4_scan<<<(B_ * N_) / 16, 256, 0, stream>>>(gx16, whh16, b_hh, tn_g, tn_b, out);
}
